// ESAMoleculeClassifier_6691559047220
// MI455X (gfx1250) — matrix operand order repaired and checked by reading, not compiled, not run on hardware
//
#include <hip/hip_runtime.h>
#include <hip/hip_bf16.h>

// ---------------------------------------------------------------------------
// Types / WMMA plumbing (CDNA5 gfx1250, wave32)
// ---------------------------------------------------------------------------
typedef unsigned int       u32;
typedef unsigned short     u16;
typedef __bf16 bf16x16 __attribute__((ext_vector_type(16)));
typedef float  f32x8   __attribute__((ext_vector_type(8)));
typedef u32    u32x4   __attribute__((ext_vector_type(4)));

union ABFrag { u32 u[8]; bf16x16 v; };

__device__ inline f32x8 wmma_bf16(const ABFrag& a, const ABFrag& b, f32x8 c) {
  return __builtin_amdgcn_wmma_f32_16x16x32_bf16(false, a.v, false, b.v,
                                                 (short)0, c, false, false);
}

// fp32 -> bf16 round-to-nearest-even
__device__ inline u16 f2bf(float f) {
  u32 u = __float_as_uint(f);
  u32 r = u + 0x7FFFu + ((u >> 16) & 1u);
  return (u16)(r >> 16);
}
// bf16 -> fp32
__device__ inline float bf2f(u16 h) {
  return __uint_as_float(((u32)h) << 16);
}

// A fragment: 16x32 bf16, source row-major (ldm in elements).
__device__ inline void load_a_frag(const u16* base, int ldm, int rowBase, u32* a) {
  const int lane = threadIdx.x & 31;
  const u16* p = base + (rowBase + (lane & 15)) * ldm;
  const int kb = (lane >> 4) * 8;
#pragma unroll
  for (int j = 0; j < 4; ++j)
    a[j] = *reinterpret_cast<const u32*>(p + kb + 2 * j);
#pragma unroll
  for (int j = 0; j < 4; ++j)
    a[4 + j] = *reinterpret_cast<const u32*>(p + 16 + kb + 2 * j);
}

// B fragment: 32(K)x16(N) bf16, source stored N-major: Bt[n][k], ld = ldn.
__device__ inline void load_b_frag(const u16* base, int ldn, int nBase, u32* b) {
  const int lane = threadIdx.x & 31;
  const u16* p = base + (nBase + (lane & 15)) * ldn + (lane >> 4) * 8;
#pragma unroll
  for (int j = 0; j < 8; ++j)
    b[j] = *reinterpret_cast<const u32*>(p + 2 * j + (j >> 2) * 8);
}

// ---------------------------------------------------------------------------
// Small utility kernels
// ---------------------------------------------------------------------------
__global__ void zero_f32_kernel(float* p, long n) {
  long i = (long)blockIdx.x * blockDim.x + threadIdx.x;
  long s = (long)gridDim.x * blockDim.x;
  for (; i < n; i += s) p[i] = 0.0f;
}

__global__ void zero_i32_kernel(int* p, int n) {
  int i = blockIdx.x * blockDim.x + threadIdx.x;
  if (i < n) p[i] = 0;
}

// src (M x Ks, fp32) -> dst (M x ldD, bf16), zero-padded cols Ks..ldD
__global__ void f32_to_bf16_pad_kernel(const float* __restrict__ src, u16* __restrict__ dst,
                                       int M, int Ks, int ldD) {
  long i = (long)blockIdx.x * blockDim.x + threadIdx.x;
  long n = (long)M * ldD;
  if (i < n) {
    int m = (int)(i / ldD), k = (int)(i - (long)m * ldD);
    dst[i] = (k < Ks) ? f2bf(src[(long)m * Ks + k]) : (u16)0;
  }
}

// W (K x N, fp32, row-major) -> Wt (N x ldWt, bf16), zero-padded cols K..ldWt
__global__ void transpose_w_kernel(const float* __restrict__ W, u16* __restrict__ Wt,
                                   int K, int N, int ldWt) {
  long i = (long)blockIdx.x * blockDim.x + threadIdx.x;
  long n = (long)N * ldWt;
  if (i < n) {
    int nn = (int)(i / ldWt), k = (int)(i - (long)nn * ldWt);
    Wt[i] = (k < K) ? f2bf(W[(long)k * N + nn]) : (u16)0;
  }
}

__global__ void count_kernel(const long long* __restrict__ ei,
                             const long long* __restrict__ batch,
                             int* __restrict__ counts, int E) {
  int i = blockIdx.x * blockDim.x + threadIdx.x;
  if (i < E) {
    int s = (int)ei[i];
    atomicAdd(&counts[(int)batch[s]], 1);
  }
}

// single block of 256 threads: exclusive scan of counts -> starts
__global__ void scan_kernel(const int* __restrict__ counts, int* __restrict__ starts, int G) {
  __shared__ int buf[256];
  int t = threadIdx.x;
  int v = (t < G) ? counts[t] : 0;
  buf[t] = v;
  __syncthreads();
  for (int off = 1; off < 256; off <<= 1) {
    int x = 0;
    if (t >= off) x = buf[t - off];
    __syncthreads();
    buf[t] += x;
    __syncthreads();
  }
  if (t < G) starts[t] = buf[t] - v;
}

__global__ void edge_idx_kernel(const long long* __restrict__ ei,
                                const long long* __restrict__ batch,
                                const int* __restrict__ starts,
                                int* __restrict__ srcIdx, int* __restrict__ dstIdx,
                                int* __restrict__ outRow, int E, int maxE) {
  int i = blockIdx.x * blockDim.x + threadIdx.x;
  if (i < E) {
    int s = (int)ei[i];
    int d = (int)ei[(long)E + i];
    int g = (int)batch[s];
    srcIdx[i] = s;
    dstIdx[i] = d;
    outRow[i] = g * maxE + (i - starts[g]);
  }
}

// ---------------------------------------------------------------------------
// bf16 WMMA GEMM:  C[M,N](bf16) = epilogue( A[M,K](bf16) @ Wt[N,K](bf16)^T )
// Block: 256 threads = 8 waves; tile 128(M) x 128(N); K-step 32.
// Wave w: rows wm=(w&3)*32 (2 A-frags), cols wn=(w>>2)*64 (4 B-frags)
//   -> 8 WMMAs per K-step.
// Staging: unconditional b128 copies from row-clamped pointers (out-of-range
// rows/cols only feed outputs discarded by the epilogue guards).
// LDS double-buffered: 1 barrier / K-step; next chunk's global loads issue
// before the WMMA sequence to hide latency behind matrix math.
// Requires K % 32 == 0 (callers pad K to 32).
// ---------------------------------------------------------------------------
#define F_RELU  1
#define F_ACC   2
#define F_RESID 4

__global__ __launch_bounds__(256) void gemm_bf16_kernel(
    const u16* __restrict__ A, int ldA,
    const u16* __restrict__ Wt, int ldW,
    const float* __restrict__ bias,
    const u16* __restrict__ resid, int ldR,
    u16* __restrict__ C, int ldC,
    int M, int N, int K,
    const int* __restrict__ gatherA,
    const int* __restrict__ scatterC,
    int flags) {
  __shared__ u16 As[2][128 * 32];
  __shared__ u16 Bs[2][128 * 32];

  const int tileM = blockIdx.y * 128;
  const int tileN = blockIdx.x * 128;
  const int t = threadIdx.x;
  const int lane = t & 31;
  const int w = t >> 5;
  const int wm = (w & 3) * 32;   // wave M offset in tile
  const int wn = (w >> 2) * 64;  // wave N offset in tile

  f32x8 acc[2][4];
#pragma unroll
  for (int a = 0; a < 2; ++a)
#pragma unroll
    for (int s = 0; s < 4; ++s)
      acc[a][s] = (f32x8){0.f, 0.f, 0.f, 0.f, 0.f, 0.f, 0.f, 0.f};

  // ---- loop-invariant staging pointers (rows clamped; loads unconditional) ----
  const int row0 = t >> 2;        // 0..63
  const int row1 = 64 + row0;     // 64..127
  const int skc  = (t & 3) * 8;   // 0,8,16,24

  const int cgr0 = min(tileM + row0, M - 1);
  const int cgr1 = min(tileM + row1, M - 1);
  const long ar0 = gatherA ? (long)gatherA[cgr0] : (long)cgr0;
  const long ar1 = gatherA ? (long)gatherA[cgr1] : (long)cgr1;
  const u16* aRow0 = A + ar0 * ldA + skc;
  const u16* aRow1 = A + ar1 * ldA + skc;

  const int cgn0 = min(tileN + row0, N - 1);
  const int cgn1 = min(tileN + row1, N - 1);
  const u16* bRow0 = Wt + (long)cgn0 * ldW + skc;
  const u16* bRow1 = Wt + (long)cgn1 * ldW + skc;

  const int sa0 = row0 * 32 + skc;
  const int sa1 = row1 * 32 + skc;

  // prologue: stage first chunk
  {
    u32x4 a0 = *reinterpret_cast<const u32x4*>(aRow0);
    u32x4 a1 = *reinterpret_cast<const u32x4*>(aRow1);
    u32x4 b0 = *reinterpret_cast<const u32x4*>(bRow0);
    u32x4 b1 = *reinterpret_cast<const u32x4*>(bRow1);
    *reinterpret_cast<u32x4*>(&As[0][sa0]) = a0;
    *reinterpret_cast<u32x4*>(&As[0][sa1]) = a1;
    *reinterpret_cast<u32x4*>(&Bs[0][sa0]) = b0;
    *reinterpret_cast<u32x4*>(&Bs[0][sa1]) = b1;
  }
  __syncthreads();

  int buf = 0;
  for (int k0 = 0; k0 < K; k0 += 32) {
    const bool more = (k0 + 32 < K);  // scalar-uniform
    u32x4 na0, na1, nb0, nb1;
    if (more) {  // prefetch next chunk (overlaps with WMMA below)
      na0 = *reinterpret_cast<const u32x4*>(aRow0 + k0 + 32);
      na1 = *reinterpret_cast<const u32x4*>(aRow1 + k0 + 32);
      nb0 = *reinterpret_cast<const u32x4*>(bRow0 + k0 + 32);
      nb1 = *reinterpret_cast<const u32x4*>(bRow1 + k0 + 32);
    }

    ABFrag af0, af1;
    load_a_frag(As[buf], 32, wm, af0.u);
    load_a_frag(As[buf], 32, wm + 16, af1.u);
#pragma unroll
    for (int s = 0; s < 4; ++s) {
      ABFrag bf;
      load_b_frag(Bs[buf], 32, wn + 16 * s, bf.u);
      acc[0][s] = wmma_bf16(af0, bf, acc[0][s]);
      acc[1][s] = wmma_bf16(af1, bf, acc[1][s]);
    }

    if (more) {  // stage next chunk into the other buffer
      *reinterpret_cast<u32x4*>(&As[buf ^ 1][sa0]) = na0;
      *reinterpret_cast<u32x4*>(&As[buf ^ 1][sa1]) = na1;
      *reinterpret_cast<u32x4*>(&Bs[buf ^ 1][sa0]) = nb0;
      *reinterpret_cast<u32x4*>(&Bs[buf ^ 1][sa1]) = nb1;
    }
    __syncthreads();
    buf ^= 1;
  }

  // epilogue: C frag layout: col n = lane&15, rows m = (lane>>4)*8 + j
  const int nn = lane & 15;
  const int mh = (lane >> 4) * 8;
#pragma unroll
  for (int a = 0; a < 2; ++a) {
#pragma unroll
    for (int j = 0; j < 8; ++j) {
      int r = tileM + wm + a * 16 + mh + j;
      if (r >= M) continue;
      long orow = scatterC ? (long)scatterC[r] : (long)r;
#pragma unroll
      for (int s = 0; s < 4; ++s) {
        int c = tileN + wn + 16 * s + nn;
        if (c >= N) continue;
        float v = acc[a][s][j];
        if (bias) v += bias[c];
        if (flags & F_RELU) v = fmaxf(v, 0.0f);
        if (flags & F_RESID) v += bf2f(resid[(long)r * ldR + c]);
        u16* p = C + orow * ldC + c;
        if (flags & F_ACC) v += bf2f(*p);
        *p = f2bf(v);
      }
    }
  }
}

// ---------------------------------------------------------------------------
// Attention kernel: one workgroup per (graph, head).  All tensors bf16.
// SAB: Nq=192 (6 waves); PMA: Nq=32 with shared (broadcast) Q (1 wave).
// Streams 192 keys in chunks of 32 with online softmax. head_dim=32 == WMMA K.
// Writes O[g*Nq + q, h*32 + d] = Qproj + softmax(QK^T/16, mask) @ V
// ---------------------------------------------------------------------------
__global__ __launch_bounds__(192) void attn_kernel(
    const u16* __restrict__ Qtok,     // [G*192, 256] (SAB) or null
    const u16* __restrict__ Kb,       // [G*192, 256]
    const u16* __restrict__ Vb,       // [G*192, 256]
    const u16* __restrict__ Qshared,  // [Nq, 256] (PMA) or null
    const int* __restrict__ counts,   // [G]
    u16* __restrict__ O,              // [G*Nq, 256]
    int Nq) {
  __shared__ u16 Qs[192 * 32];
  __shared__ u16 Ks[192 * 32];    // key-major: Ks[key][d]  (WMMA B layout, N=key)
  __shared__ u16 Vt[32 * 192];    // hd-major:  Vt[d][key]  (WMMA B layout, N=d)
  __shared__ u16 Ps[6 * 16 * 32]; // per-wave P scratch (C-frag -> A-frag round trip)

  const int g = blockIdx.x;
  const int h = blockIdx.y;
  const int cnt = counts[g];
  const long tokBase = (long)g * 192;
  const bool pma = (Qshared != nullptr);

  // stage K (pure b128 copy) and V^T (transposed scatter)
  for (int i = threadIdx.x; i < 192 * 4; i += blockDim.x) {
    int key = i >> 2, dc = (i & 3) * 8;
    long src = (tokBase + key) * 256 + h * 32 + dc;
    *reinterpret_cast<u32x4*>(&Ks[key * 32 + dc]) =
        *reinterpret_cast<const u32x4*>(Kb + src);
    u32x4 vv = *reinterpret_cast<const u32x4*>(Vb + src);
    const u16* vs = reinterpret_cast<const u16*>(&vv);
#pragma unroll
    for (int j = 0; j < 8; ++j) Vt[(dc + j) * 192 + key] = vs[j];
  }
  // stage Q (pure b128 copy)
  for (int i = threadIdx.x; i < Nq * 4; i += blockDim.x) {
    int q = i >> 2, dc = (i & 3) * 8;
    const u16* qp = (pma ? Qshared + (long)q * 256 : Qtok + (tokBase + q) * 256)
                    + h * 32 + dc;
    *reinterpret_cast<u32x4*>(&Qs[q * 32 + dc]) =
        *reinterpret_cast<const u32x4*>(qp);
  }
  __syncthreads();

  const int w = threadIdx.x >> 5;
  const int lane = threadIdx.x & 31;
  const int nn = lane & 15;
  u16* Pw = Ps + w * (16 * 32);
  const float scale = 0.0625f;  // 1/sqrt(256)

  for (int qt = 0; qt < 2; ++qt) {
    const int qbase = w * 32 + qt * 16;
    ABFrag qa;
    load_a_frag(Qs, 32, qbase, qa.u);

    f32x8 o0 = {0.f, 0.f, 0.f, 0.f, 0.f, 0.f, 0.f, 0.f};
    f32x8 o1 = {0.f, 0.f, 0.f, 0.f, 0.f, 0.f, 0.f, 0.f};
    float runmax[8], runsum[8];
#pragma unroll
    for (int j = 0; j < 8; ++j) { runmax[j] = -1e30f; runsum[j] = 0.0f; }

    for (int kk = 0; kk < 192; kk += 32) {
      ABFrag kb0, kb1;
      load_b_frag(Ks, 32, kk, kb0.u);
      load_b_frag(Ks, 32, kk + 16, kb1.u);
      f32x8 s0 = {0.f, 0.f, 0.f, 0.f, 0.f, 0.f, 0.f, 0.f};
      f32x8 s1 = {0.f, 0.f, 0.f, 0.f, 0.f, 0.f, 0.f, 0.f};
      s0 = wmma_bf16(qa, kb0, s0);
      s1 = wmma_bf16(qa, kb1, s1);

      const int key0 = kk + nn, key1 = kk + 16 + nn;
#pragma unroll
      for (int j = 0; j < 8; ++j) {
        float a = s0[j] * scale;
        float b = s1[j] * scale;
        if (key0 >= cnt) a = -1e9f;
        if (key1 >= cnt) b = -1e9f;
        s0[j] = a;
        s1[j] = b;
      }
      // online softmax update (row reductions across 16-lane half-groups)
#pragma unroll
      for (int j = 0; j < 8; ++j) {
        float m = fmaxf(s0[j], s1[j]);
#pragma unroll
        for (int off = 1; off < 16; off <<= 1) m = fmaxf(m, __shfl_xor(m, off, 32));
        float nm = fmaxf(runmax[j], m);
        float alpha = __expf(runmax[j] - nm);
        runmax[j] = nm;
        o0[j] *= alpha;
        o1[j] *= alpha;
        float p0 = __expf(s0[j] - nm);
        float p1 = __expf(s1[j] - nm);
        s0[j] = p0;
        s1[j] = p1;
        float rs = p0 + p1;
#pragma unroll
        for (int off = 1; off < 16; off <<= 1) rs += __shfl_xor(rs, off, 32);
        runsum[j] = runsum[j] * alpha + rs;
      }
      // C-frag -> A-frag relayout via per-wave LDS scratch (same-wave ordering)
#pragma unroll
      for (int j = 0; j < 8; ++j) {
        int r = ((lane >> 4) << 3) + j;
        Pw[r * 32 + nn] = f2bf(s0[j]);
        Pw[r * 32 + 16 + nn] = f2bf(s1[j]);
      }
      ABFrag pa, vb0, vb1;
      load_a_frag(Pw, 32, 0, pa.u);
      load_b_frag(Vt + kk, 192, 0, vb0.u);    // hd 0..15
      load_b_frag(Vt + kk, 192, 16, vb1.u);   // hd 16..31
      o0 = wmma_bf16(pa, vb0, o0);
      o1 = wmma_bf16(pa, vb1, o1);
    }

    // epilogue: normalize + add projected-Q residual
#pragma unroll
    for (int j = 0; j < 8; ++j) {
      int r = qbase + ((lane >> 4) << 3) + j;
      float inv = 1.0f / runsum[j];
      int c0 = h * 32 + nn;
      int c1 = c0 + 16;
      float q0 = bf2f(pma ? Qshared[(long)r * 256 + c0] : Qtok[(tokBase + r) * 256 + c0]);
      float q1 = bf2f(pma ? Qshared[(long)r * 256 + c1] : Qtok[(tokBase + r) * 256 + c1]);
      long orow = (long)g * Nq + r;
      O[orow * 256 + c0] = f2bf(q0 + o0[j] * inv);
      O[orow * 256 + c1] = f2bf(q1 + o1[j] * inv);
    }
  }
}

// logits[g] = b2 + sum_n hc[g,n] * w2[n]   (one wave per graph)
__global__ void logits_kernel(const u16* __restrict__ hc, const float* __restrict__ w2,
                              const float* __restrict__ b2, float* __restrict__ out) {
  int g = blockIdx.x;
  int lane = threadIdx.x;
  float s = 0.0f;
  for (int n = lane; n < 256; n += 32) s += bf2f(hc[(long)g * 256 + n]) * w2[n];
#pragma unroll
  for (int off = 16; off >= 1; off >>= 1) s += __shfl_xor(s, off, 32);
  if (lane == 0) out[g] = s + b2[0];
}

// ---------------------------------------------------------------------------
// Host orchestration
// ---------------------------------------------------------------------------
enum {
  IN_X = 0, IN_EDGE_ATTR = 1,
  IN_NODE_W = 2, IN_NODE_B = 3,
  IN_EDGE_W = 4, IN_EDGE_B = 5,
  IN_ESET_W = 6, IN_ESET_B = 7,
  IN_SAB0 = 8,    // wq,bq,wk,bk,wv,bv,wo,bo
  IN_SAB1 = 16,
  IN_SAB2 = 24,
  IN_PMA = 32,    // wq,bq,wk,bk,wv,bv,wo,bo
  IN_PMA_SEEDS = 40,
  IN_CLS_W1 = 41, IN_CLS_B1 = 42, IN_CLS_W2 = 43, IN_CLS_B2 = 44,
  IN_EDGE_INDEX = 45, IN_BATCH = 46, IN_NUM_GRAPHS = 47, IN_MAX_EDGES = 48,
};

extern "C" void kernel_launch(void* const* d_in, const int* in_sizes, int n_in,
                              void* d_out, int out_size, void* d_ws, size_t ws_size,
                              hipStream_t stream) {
  const int Hd = 256, NHh = 8, NIi = 32, ME = 192, NPG = 64;
  const int G = in_sizes[IN_BATCH] / NPG;          // 256
  const int Etot = in_sizes[IN_EDGE_ATTR] / 16;    // ~32641
  const int Ntok = G * ME;                          // 49152
  const int Nnodes = in_sizes[IN_X] / 32;          // 16384

  char* base = (char*)d_ws;
  size_t off = 0;
  auto alloc = [&](size_t bytes) -> void* {
    void* p = base + off;
    off += (bytes + 255) & ~(size_t)255;
    return p;
  };

  // bf16 transposed weights (edge padded to K=32)
  u16* WtNode = (u16*)alloc((size_t)Hd * 32 * 2);
  u16* WtEdge = (u16*)alloc((size_t)Hd * 32 * 2);   // 16 -> padded 32
  u16* WtEset = (u16*)alloc((size_t)Hd * 768 * 2);
  u16* WtMab[4][4];
  for (int l = 0; l < 4; ++l)
    for (int m = 0; m < 4; ++m) WtMab[l][m] = (u16*)alloc((size_t)Hd * Hd * 2);
  u16* WtCls1 = (u16*)alloc((size_t)Hd * (NIi * Hd) * 2);

  // bf16 converted inputs (edge_attr padded to 32 cols)
  u16* xbf    = (u16*)alloc((size_t)Nnodes * 32 * 2);
  u16* eabf   = (u16*)alloc((size_t)Etot * 32 * 2);
  u16* seedbf = (u16*)alloc((size_t)NIi * Hd * 2);

  // bf16 activations
  u16* hbuf   = (u16*)alloc((size_t)Nnodes * Hd * 2);
  u16* ebuf   = (u16*)alloc((size_t)Etot * Hd * 2);
  u16* Ebuf   = (u16*)alloc((size_t)Ntok * Hd * 2);
  u16* Qb     = (u16*)alloc((size_t)Ntok * Hd * 2);
  u16* Kb     = (u16*)alloc((size_t)Ntok * Hd * 2);
  u16* Vb     = (u16*)alloc((size_t)Ntok * Hd * 2);
  u16* Ob     = (u16*)alloc((size_t)Ntok * Hd * 2);
  u16* Qp     = (u16*)alloc((size_t)NIi * Hd * 2);
  u16* poolO  = (u16*)alloc((size_t)G * NIi * Hd * 2);
  u16* pooled = (u16*)alloc((size_t)G * NIi * Hd * 2);
  u16* hc     = (u16*)alloc((size_t)G * Hd * 2);
  int* counts = (int*)alloc((size_t)G * 4);
  int* starts = (int*)alloc((size_t)G * 4);
  int* srcIdx = (int*)alloc((size_t)Etot * 4);
  int* dstIdx = (int*)alloc((size_t)Etot * 4);
  int* outRow = (int*)alloc((size_t)Etot * 4);

  // --- weight convert/transpose (optionally K-padded) ---
  auto txw = [&](int idx, u16* dst, int K, int N, int ldWt) {
    long tot = (long)N * ldWt;
    transpose_w_kernel<<<dim3((int)((tot + 255) / 256)), dim3(256), 0, stream>>>(
        (const float*)d_in[idx], dst, K, N, ldWt);
  };
  txw(IN_NODE_W, WtNode, 32, Hd, 32);
  txw(IN_EDGE_W, WtEdge, 16, Hd, 32);   // zero-padded K 16 -> 32
  txw(IN_ESET_W, WtEset, 768, Hd, 768);
  const int mabBase[4] = {IN_SAB0, IN_SAB1, IN_SAB2, IN_PMA};
  for (int l = 0; l < 4; ++l)
    for (int m = 0; m < 4; ++m) txw(mabBase[l] + 2 * m, WtMab[l][m], Hd, Hd, Hd);
  txw(IN_CLS_W1, WtCls1, NIi * Hd, Hd, NIi * Hd);

  // --- convert fp32 inputs to bf16 (edge_attr zero-padded 16 -> 32 cols) ---
  auto cvt = [&](int idx, u16* dst, int M, int Ks, int ldD) {
    long tot = (long)M * ldD;
    f32_to_bf16_pad_kernel<<<dim3((int)((tot + 255) / 256)), dim3(256), 0, stream>>>(
        (const float*)d_in[idx], dst, M, Ks, ldD);
  };
  cvt(IN_X, xbf, Nnodes, 32, 32);
  cvt(IN_EDGE_ATTR, eabf, Etot, 16, 32);
  cvt(IN_PMA_SEEDS, seedbf, NIi, Hd, Hd);

  // --- edge bookkeeping (counts / starts / gather-scatter indices) ---
  const long long* ei = (const long long*)d_in[IN_EDGE_INDEX];
  const long long* batch = (const long long*)d_in[IN_BATCH];
  zero_i32_kernel<<<dim3(1), dim3(256), 0, stream>>>(counts, G);
  count_kernel<<<dim3((Etot + 255) / 256), dim3(256), 0, stream>>>(ei, batch, counts, Etot);
  scan_kernel<<<dim3(1), dim3(256), 0, stream>>>(counts, starts, G);
  edge_idx_kernel<<<dim3((Etot + 255) / 256), dim3(256), 0, stream>>>(
      ei, batch, starts, srcIdx, dstIdx, outRow, Etot, ME);

  auto gemm = [&](const u16* A, int ldA, const u16* Wt, int ldW, const float* bias,
                  const u16* resid, int ldR, u16* C, int ldC, int M, int N, int K,
                  const int* gA, const int* sC, int flags) {
    dim3 grid((N + 127) / 128, (M + 127) / 128);
    gemm_bf16_kernel<<<grid, dim3(256), 0, stream>>>(A, ldA, Wt, ldW, bias, resid, ldR,
                                                     C, ldC, M, N, K, gA, sC, flags);
  };

  // --- node / edge embeddings ---
  gemm(xbf, 32, WtNode, 32, (const float*)d_in[IN_NODE_B],
       nullptr, 0, hbuf, Hd, Nnodes, Hd, 32, nullptr, nullptr, 0);
  gemm(eabf, 32, WtEdge, 32, (const float*)d_in[IN_EDGE_B],
       nullptr, 0, ebuf, Hd, Etot, Hd, 32, nullptr, nullptr, 0);

  // --- edge_feats = [h[src]|h[dst]|e] @ Weset + b, scattered into E[g, pos] ---
  zero_f32_kernel<<<dim3(2048), dim3(256), 0, stream>>>((float*)Ebuf,
                                                        (long)Ntok * Hd / 2);
  gemm(hbuf, Hd, WtEset, 768, nullptr, nullptr, 0, Ebuf, Hd, Etot, Hd, Hd,
       srcIdx, outRow, 0);
  gemm(hbuf, Hd, WtEset + 256, 768, nullptr, nullptr, 0, Ebuf, Hd, Etot, Hd, Hd,
       dstIdx, outRow, F_ACC);
  gemm(ebuf, Hd, WtEset + 512, 768, (const float*)d_in[IN_ESET_B], nullptr, 0,
       Ebuf, Hd, Etot, Hd, Hd, nullptr, outRow, F_ACC);

  // --- 3 SAB layers ---
  for (int l = 0; l < 3; ++l) {
    const float* bq = (const float*)d_in[mabBase[l] + 1];
    const float* bk = (const float*)d_in[mabBase[l] + 3];
    const float* bv = (const float*)d_in[mabBase[l] + 5];
    const float* bo = (const float*)d_in[mabBase[l] + 7];
    gemm(Ebuf, Hd, WtMab[l][0], Hd, bq, nullptr, 0, Qb, Hd, Ntok, Hd, Hd, nullptr, nullptr, 0);
    gemm(Ebuf, Hd, WtMab[l][1], Hd, bk, nullptr, 0, Kb, Hd, Ntok, Hd, Hd, nullptr, nullptr, 0);
    gemm(Ebuf, Hd, WtMab[l][2], Hd, bv, nullptr, 0, Vb, Hd, Ntok, Hd, Hd, nullptr, nullptr, 0);
    attn_kernel<<<dim3(G, NHh), dim3(192), 0, stream>>>(Qb, Kb, Vb, nullptr, counts, Ob, ME);
    // E = O + relu(O @ Wo + bo)
    gemm(Ob, Hd, WtMab[l][3], Hd, bo, Ob, Hd, Ebuf, Hd, Ntok, Hd, Hd,
         nullptr, nullptr, F_RELU | F_RESID);
  }

  // --- PMA ---
  {
    const int l = 3;
    const float* bq = (const float*)d_in[mabBase[l] + 1];
    const float* bk = (const float*)d_in[mabBase[l] + 3];
    const float* bv = (const float*)d_in[mabBase[l] + 5];
    const float* bo = (const float*)d_in[mabBase[l] + 7];
    gemm(seedbf, Hd, WtMab[l][0], Hd, bq, nullptr, 0,
         Qp, Hd, NIi, Hd, Hd, nullptr, nullptr, 0);
    gemm(Ebuf, Hd, WtMab[l][1], Hd, bk, nullptr, 0, Kb, Hd, Ntok, Hd, Hd, nullptr, nullptr, 0);
    gemm(Ebuf, Hd, WtMab[l][2], Hd, bv, nullptr, 0, Vb, Hd, Ntok, Hd, Hd, nullptr, nullptr, 0);
    attn_kernel<<<dim3(G, NHh), dim3(32), 0, stream>>>(nullptr, Kb, Vb, Qp, counts, poolO, NIi);
    gemm(poolO, Hd, WtMab[l][3], Hd, bo, poolO, Hd, pooled, Hd, G * NIi, Hd, Hd,
         nullptr, nullptr, F_RELU | F_RESID);
  }

  // --- classifier ---
  gemm(pooled, NIi * Hd, WtCls1, NIi * Hd, (const float*)d_in[IN_CLS_B1], nullptr, 0,
       hc, Hd, G, Hd, NIi * Hd, nullptr, nullptr, F_RELU);
  logits_kernel<<<dim3(G), dim3(32), 0, stream>>>(
      hc, (const float*)d_in[IN_CLS_W2], (const float*)d_in[IN_CLS_B2], (float*)d_out);
}
